// MoE_39032662786518
// MI455X (gfx1250) — matrix operand order repaired and checked by reading, not compiled, not run on hardware
//
#include <hip/hip_runtime.h>
#include <hip/hip_bf16.h>
#include <hip/hip_fp16.h>

// ---------------------------------------------------------------------------
// MI455X (gfx1250): implicit-GEMM conv + classifier via v_wmma_f32_16x16x32_f16,
// wave32, 16x64 block per wave, fully-unrolled ping-pong k-loop with sched
// fences to keep prefetch loads ahead of WMMAs, TDM (tensor_load_to_lds)
// staging of shared weight blocks into LDS.
// ---------------------------------------------------------------------------

typedef _Float16 half_t;
typedef __attribute__((ext_vector_type(16))) _Float16 v16h;
typedef __attribute__((ext_vector_type(8)))  float    v8f;
typedef __attribute__((ext_vector_type(4))) unsigned int u32x4;
typedef __attribute__((ext_vector_type(8))) unsigned int u32x8;

union AFrag { v16h v; uint4 q[2]; half_t h[16]; };
union H8    { uint4 q; half_t h[8]; };

#define BATCH   256
#define EXPERTS 16
#define NCLS    1000
#define TCH     256
#define CAPSZ   64

#ifndef USE_TDM
#define USE_TDM 1
#endif

#if __has_builtin(__builtin_amdgcn_sched_barrier)
#define SCHED_FENCE() __builtin_amdgcn_sched_barrier(0)
#else
#define SCHED_FENCE() ((void)0)
#endif

// ---------------------------------------------------------------------------
// Weight packing: fragment order out[((kt*ntiles+nt)*32 + lane)*16 + h]
// B-matrix 32x16 f16 lane layout: lane holds column N = lane%16,
// half h holds K = (h&7) + (h>=8?16:0) + (lane>=16?8:0).  k-order: k = tap*Ci + ci.
// ---------------------------------------------------------------------------
__global__ void pack_conv_weights(const float* __restrict__ w, half_t* __restrict__ out,
                                  int Ci, int Co, int Ktot, int Kpad) {
    int idx = blockIdx.x * blockDim.x + threadIdx.x;
    int total = (Kpad >> 5) * (Co >> 4) * 512;
    if (idx >= total) return;
    int h    = idx & 15;
    int lane = (idx >> 4) & 31;
    int frag = idx >> 9;
    int ntiles = Co >> 4;
    int kt = frag / ntiles;
    int nt = frag - kt * ntiles;
    int kk = ((h >> 3) << 4) + ((lane >> 4) << 3) + (h & 7);
    int kg = kt * 32 + kk;
    int co = nt * 16 + (lane & 15);
    half_t val = (half_t)0.f;
    if (kg < Ktot) {
        int tap = kg / Ci;
        int ci  = kg - tap * Ci;
        int ky = tap / 3, kx = tap - ky * 3;
        val = (half_t)w[(((size_t)co * Ci + ci) * 3 + ky) * 3 + kx];
    }
    out[idx] = val;
}

__global__ void pack_cls_weights(const float* __restrict__ w /*[E][NC][TC]*/,
                                 half_t* __restrict__ out) {
    int idx = blockIdx.x * blockDim.x + threadIdx.x;
    int total = 8 * 1000 * 512;
    if (idx >= total) return;
    int h    = idx & 15;
    int lane = (idx >> 4) & 31;
    int frag = idx >> 9;
    int kt = frag / 1000;
    int nt = frag - kt * 1000;
    int k  = kt * 32 + ((h >> 3) << 4) + ((lane >> 4) << 3) + (h & 7);
    int n  = nt * 16 + (lane & 15);
    int e  = n / NCLS, c = n - e * NCLS;
    out[idx] = (half_t)w[((size_t)e * NCLS + c) * TCH + k];
}

__global__ void nchw_to_nhwc_f16(const float* __restrict__ x, half_t* __restrict__ out,
                                 int C, int H, int W) {
    int idx = blockIdx.x * blockDim.x + threadIdx.x;
    int total = BATCH * C * H * W;
    if (idx >= total) return;
    int c = idx % C;
    int t = idx / C;
    int px = t % W; t /= W;
    int py = t % H; t /= H;
    int b  = t;
    out[idx] = (half_t)x[(((size_t)b * C + c) * H + py) * W + px];
}

// ---------------------------------------------------------------------------
// TDM: stage `nhalves` f16 words from global into LDS with one descriptor.
// D# per CDNA5 ISA ch.8: group0 = {count/flags, lds_addr, global_addr, type=2},
// group1 = {mask/data_size, dims}: 1 row x tile_dim0=N elements, data_size=2B.
// Issued by one wave; completion via s_wait_tensorcnt then workgroup barrier.
// ---------------------------------------------------------------------------
__device__ __forceinline__ void tdm_stage_to_lds(const half_t* gsrc, half_t* lds_dst,
                                                 unsigned nhalves) {
#if USE_TDM && __has_builtin(__builtin_amdgcn_s_wait_tensorcnt)
    if (threadIdx.x < 32) {
        unsigned long long ga = (unsigned long long)(uintptr_t)gsrc;
        unsigned lad = (unsigned)(uintptr_t)lds_dst;   // low 32 bits = LDS byte address
        unsigned N = nhalves;
        u32x4 g0;
        g0[0] = 1u;                                    // count=1, is_restore=0
        g0[1] = lad;                                   // lds_addr
        g0[2] = (unsigned)ga;                          // global_addr[31:0]
        g0[3] = (unsigned)(ga >> 32) | (2u << 30);     // global_addr[56:32] | type=2
        u32x8 g1;
        g1[0] = 1u << 16;                              // workgroup_mask=0, data_size=1 (2B)
        g1[1] = (N & 0xffffu) << 16;                   // tensor_dim0[15:0]
        g1[2] = (N >> 16) | (1u << 16);                // tensor_dim0[31:16] | tensor_dim1=1
        g1[3] = (N & 0xffffu) << 16;                   // tile_dim0 = N
        g1[4] = 1u;                                    // tile_dim1 = 1
        g1[5] = N;                                     // tensor_dim0_stride[31:0]
        g1[6] = 0u;
        g1[7] = 0u;
        asm volatile("tensor_load_to_lds %0, %1" :: "s"(g0), "s"(g1) : "memory");
        __builtin_amdgcn_s_wait_tensorcnt((short)0);
    }
#else
    for (unsigned i = threadIdx.x * 8; i < nhalves; i += blockDim.x * 8)
        *(uint4*)(lds_dst + i) = *(const uint4*)(gsrc + i);
#endif
    __syncthreads();
}

// ---------------------------------------------------------------------------
// Implicit-GEMM 3x3 SAME conv, fused BN(eval)+ReLU.
// One wave -> 16 pixels x (16*NW) channels; fully-unrolled ping-pong k-loop.
// A 16x32 f16 lane layout: lane holds row M=lane%16, half h holds
// K = (h&7) + (h>=8?16:0) + (lane>=16?8:0)  => two contiguous 8-half runs.
// STAGE: whole packed weight block staged to LDS via TDM (ngroups==1 layers).
// ---------------------------------------------------------------------------
template<int CI, int CO, int NW, bool STAGE>
__global__ void __launch_bounds__(256)
conv3x3_wmma(const half_t* __restrict__ in, const half_t* __restrict__ wpack,
             const float* __restrict__ cb, const float* __restrict__ bg,
             const float* __restrict__ bb, half_t* __restrict__ out,
             int H, int W) {
    constexpr int KSTEPS = (CI >= 32) ? (9 * CI) / 32 : 1;
    constexpr int NTILES = CO / 16;
    constexpr int NGROUPS = NTILES / NW;
    __shared__ __align__(64) half_t smw[STAGE ? (KSTEPS * NTILES * 512) : 64];

    if constexpr (STAGE)
        tdm_stage_to_lds(wpack, smw, (unsigned)(KSTEPS * NTILES * 512));

    const int lane = threadIdx.x & 31;
    const int gid  = (blockIdx.x * blockDim.x + threadIdx.x) >> 5;
    const int mtile = gid / NGROUPS;
    const int ng    = gid - mtile * NGROUPS;
    const int HW = H * W;
    const int Mtiles = (BATCH * HW) >> 4;
    if (mtile >= Mtiles) return;

    const int m  = mtile * 16 + (lane & 15);
    const int b  = m / HW;
    const int r0 = m - b * HW;
    const int oy = r0 / W;
    const int ox = r0 - oy * W;
    const int off = (lane >> 4) << 3;

    auto loadA = [&](int s, AFrag& a) {
        if constexpr (CI >= 32) {
            const int taps_per = CI / 32;
            const int tap = s / taps_per;
            const int cib = (s - tap * taps_per) * 32;
            const int ky = tap / 3, kx = tap - ky * 3;
            const int iy = oy + ky - 1, ix = ox + kx - 1;
            if ((unsigned)iy < (unsigned)H && (unsigned)ix < (unsigned)W) {
                const half_t* p = in + ((size_t)(b * H + iy) * W + ix) * CI + cib + off;
                a.q[0] = *(const uint4*)(p);
                a.q[1] = *(const uint4*)(p + 16);
            } else {
                a.q[0] = make_uint4(0u,0u,0u,0u);
                a.q[1] = make_uint4(0u,0u,0u,0u);
            }
        } else {
            #pragma unroll
            for (int h = 0; h < 16; ++h) {
                int K = (h & 7) + ((h >> 3) << 4) + off;
                half_t v = (half_t)0.f;
                if (K < 9 * CI) {
                    int tap = K / CI;
                    int ci  = K - tap * CI;
                    int ky = tap / 3, kx = tap - ky * 3;
                    int iy = oy + ky - 1, ix = ox + kx - 1;
                    if ((unsigned)iy < (unsigned)H && (unsigned)ix < (unsigned)W)
                        v = in[((size_t)(b * H + iy) * W + ix) * CI + ci];
                }
                a.h[h] = v;
            }
        }
    };
    auto loadB = [&](int s, int nw) -> v16h {
        const size_t idx = (((size_t)s * NTILES + (ng * NW + nw)) * 32 + lane) * 16;
        if constexpr (STAGE) return *(const v16h*)(smw + idx);
        else                 return *(const v16h*)(wpack + idx);
    };

    v8f acc[NW];
    #pragma unroll
    for (int i = 0; i < NW; ++i) acc[i] = (v8f){0.f,0.f,0.f,0.f,0.f,0.f,0.f,0.f};

    // fully-unrolled software pipeline: ping-pong fragment buffers by parity,
    // step s+1 loads fenced ahead of step s WMMAs.
    AFrag a[2]; v16h bf[2][NW];
    loadA(0, a[0]);
    #pragma unroll
    for (int nw = 0; nw < NW; ++nw) bf[0][nw] = loadB(0, nw);

    #pragma unroll
    for (int s = 0; s < KSTEPS; ++s) {
        const int cur = s & 1, nxt = cur ^ 1;
        if (s + 1 < KSTEPS) {
            loadA(s + 1, a[nxt]);
            #pragma unroll
            for (int nw = 0; nw < NW; ++nw) bf[nxt][nw] = loadB(s + 1, nw);
        }
        SCHED_FENCE();   // keep s+1 loads issued before s WMMAs -> partial waits
        #pragma unroll
        for (int nw = 0; nw < NW; ++nw)
            acc[nw] = __builtin_amdgcn_wmma_f32_16x16x32_f16(
                false, a[cur].v, false, bf[cur][nw], (short)0, acc[nw], false, false);
        SCHED_FENCE();
    }

    // fused BN (running stats trivial) + bias + ReLU -> f16 NHWC
    const float rs = 0.9999950000374997f;        // 1/sqrt(1+1e-5)
    const int hi = (lane >> 4) << 3;
    #pragma unroll
    for (int nw = 0; nw < NW; ++nw) {
        const int n = (ng * NW + nw) * 16 + (lane & 15);
        const float scale = bg[n] * rs;
        const float bias  = cb[n] * scale + bb[n];
        #pragma unroll
        for (int r = 0; r < 8; ++r) {
            float v = acc[nw][r] * scale + bias;
            v = v > 0.f ? v : 0.f;
            out[(size_t)(mtile * 16 + r + hi) * CO + n] = (half_t)v;
        }
    }
}

// 2x2 maxpool, NHWC f16, vectorized 8 channels / thread (b128 path)
__global__ void maxpool2x2_nhwc_v8(const half_t* __restrict__ in, half_t* __restrict__ out,
                                   int H, int W, int C) {
    int idx = blockIdx.x * blockDim.x + threadIdx.x;
    int Ho = H >> 1, Wo = W >> 1;
    int c8 = C >> 3;
    int total = BATCH * Ho * Wo * c8;
    if (idx >= total) return;
    int cg = idx % c8;
    int t  = idx / c8;
    int px = t % Wo; t /= Wo;
    int py = t % Ho; t /= Ho;
    int b  = t;
    int c  = cg * 8;
    const half_t* p = in + ((size_t)(b * H + py * 2) * W + px * 2) * C + c;
    H8 a0, a1, a2, a3, o;
    a0.q = *(const uint4*)(p);
    a1.q = *(const uint4*)(p + C);
    a2.q = *(const uint4*)(p + (size_t)W * C);
    a3.q = *(const uint4*)(p + (size_t)W * C + C);
    #pragma unroll
    for (int i = 0; i < 8; ++i) {
        float m = fmaxf(fmaxf((float)a0.h[i], (float)a1.h[i]),
                        fmaxf((float)a2.h[i], (float)a3.h[i]));
        o.h[i] = (half_t)m;
    }
    *(uint4*)(out + ((size_t)(b * Ho + py) * Wo + px) * C + c) = o.q;
}

__global__ void avgpool_feats(const half_t* __restrict__ in, half_t* __restrict__ feats) {
    int idx = blockIdx.x * blockDim.x + threadIdx.x;   // 256*256, coalesced over c
    if (idx >= BATCH * TCH) return;
    int b = idx >> 8, c = idx & 255;
    const half_t* p = in + (size_t)b * 16 * 16 * 256 + c;
    float s = 0.f;
    for (int i = 0; i < 256; ++i) s += (float)p[(size_t)i * 256];
    feats[idx] = (half_t)(s * (1.f / 256.f));
}

// Classifier: [256 x 16000] = feats[256 x 256] * Wt[256 x 16000] + bias
__global__ void __launch_bounds__(256)
cls_gemm_wmma(const half_t* __restrict__ feats, const half_t* __restrict__ wpack,
              const float* __restrict__ clsb, float* __restrict__ logits) {
    constexpr int NW = 4;
    constexpr int KS = 8;
    const int lane = threadIdx.x & 31;
    const int gid  = (blockIdx.x * blockDim.x + threadIdx.x) >> 5;
    const int ngroups = 250;
    const int mtile = gid / ngroups;
    const int ng    = gid - mtile * ngroups;
    if (mtile >= 16) return;

    const int m   = mtile * 16 + (lane & 15);
    const int off = (lane >> 4) << 3;

    auto loadA = [&](int s, AFrag& a) {
        const half_t* p = feats + (size_t)m * TCH + s * 32 + off;
        a.q[0] = *(const uint4*)(p);
        a.q[1] = *(const uint4*)(p + 16);
    };
    auto loadB = [&](int s, int nw) -> v16h {
        return *(const v16h*)(wpack + (((size_t)s * 1000 + ng * NW + nw) * 32 + lane) * 16);
    };

    v8f acc[NW];
    #pragma unroll
    for (int i = 0; i < NW; ++i) acc[i] = (v8f){0.f,0.f,0.f,0.f,0.f,0.f,0.f,0.f};

    AFrag a[2]; v16h bf[2][NW];
    loadA(0, a[0]);
    #pragma unroll
    for (int nw = 0; nw < NW; ++nw) bf[0][nw] = loadB(0, nw);

    #pragma unroll
    for (int s = 0; s < KS; ++s) {
        const int cur = s & 1, nxt = cur ^ 1;
        if (s + 1 < KS) {
            loadA(s + 1, a[nxt]);
            #pragma unroll
            for (int nw = 0; nw < NW; ++nw) bf[nxt][nw] = loadB(s + 1, nw);
        }
        SCHED_FENCE();   // keep s+1 loads issued before s WMMAs -> partial waits
        #pragma unroll
        for (int nw = 0; nw < NW; ++nw)
            acc[nw] = __builtin_amdgcn_wmma_f32_16x16x32_f16(
                false, a[cur].v, false, bf[cur][nw], (short)0, acc[nw], false, false);
        SCHED_FENCE();
    }

    const int hi = (lane >> 4) << 3;
    #pragma unroll
    for (int nw = 0; nw < NW; ++nw) {
        const int n = (ng * NW + nw) * 16 + (lane & 15);
        const int e = n / NCLS, c = n - e * NCLS;
        const float bias = clsb[e * NCLS + c];
        #pragma unroll
        for (int r = 0; r < 8; ++r)
            logits[(size_t)(mtile * 16 + r + hi) * (EXPERTS * NCLS) + n] = acc[nw][r] + bias;
    }
}

// conf = -entropy of softmax over NC, per (b,e)
__global__ void softmax_entropy(const float* __restrict__ logits,
                                float* __restrict__ conf, float* __restrict__ out_conf) {
    __shared__ float red[256];
    const int be = blockIdx.x;
    const int t  = threadIdx.x;
    const float* l = logits + (size_t)be * NCLS;
    float mx = -1e30f;
    for (int i = t; i < NCLS; i += 256) mx = fmaxf(mx, l[i]);
    red[t] = mx; __syncthreads();
    for (int s = 128; s > 0; s >>= 1) { if (t < s) red[t] = fmaxf(red[t], red[t + s]); __syncthreads(); }
    mx = red[0]; __syncthreads();
    float s1 = 0.f, s2 = 0.f;
    for (int i = t; i < NCLS; i += 256) { float d = l[i] - mx; float e = __expf(d); s1 += e; s2 += e * d; }
    red[t] = s1; __syncthreads();
    for (int s = 128; s > 0; s >>= 1) { if (t < s) red[t] += red[t + s]; __syncthreads(); }
    s1 = red[0]; __syncthreads();
    red[t] = s2; __syncthreads();
    for (int s = 128; s > 0; s >>= 1) { if (t < s) red[t] += red[t + s]; __syncthreads(); }
    s2 = red[0];
    if (t == 0) {
        float cf = s2 / s1 - __logf(s1);
        conf[be] = cf;
        out_conf[be] = cf;
    }
}

// Capacity top-CAP over batch per expert (rank with index tie-break).
__global__ void cap_route(const float* __restrict__ conf, int* __restrict__ D1) {
    __shared__ float cf[BATCH];
    const int e = blockIdx.x;
    const int b = threadIdx.x;
    cf[b] = conf[b * EXPERTS + e];
    __syncthreads();
    const float c = cf[b];
    int rank = 0;
    for (int j = 0; j < BATCH; ++j) {
        float cj = cf[j];
        rank += (cj > c) || (cj == c && j < b);
    }
    D1[b * EXPERTS + e] = (rank < CAPSZ) ? 1 : 0;
}

// Per-sample top-2 over experts (masked conf); exactly 2 set per row.
__global__ void sample_route(const float* __restrict__ conf, const int* __restrict__ D1,
                             float* __restrict__ outD, int* __restrict__ D2,
                             float* __restrict__ nrm) {
    int b = blockIdx.x * blockDim.x + threadIdx.x;
    if (b >= BATCH) return;
    float best1 = -1e38f; int i1 = 0;
    for (int e = 0; e < EXPERTS; ++e) {
        float v = D1[b * EXPERTS + e] ? conf[b * EXPERTS + e] : -1e9f;
        if (v > best1) { best1 = v; i1 = e; }
    }
    float best2 = -1e38f; int i2 = 0;
    for (int e = 0; e < EXPERTS; ++e) {
        if (e == i1) continue;
        float v = D1[b * EXPERTS + e] ? conf[b * EXPERTS + e] : -1e9f;
        if (v > best2) { best2 = v; i2 = e; }
    }
    int cnt = 0;
    for (int e = 0; e < EXPERTS; ++e) {
        int d = (e == i1) || (e == i2);
        D2[b * EXPERTS + e] = d;
        outD[b * EXPERTS + e] = (float)d;
        cnt += d;
    }
    nrm[b] = (float)(cnt > 0 ? cnt : 1);
}

__global__ void combine(const float* __restrict__ logits, const float* __restrict__ conf,
                        const int* __restrict__ D2, const float* __restrict__ nrm,
                        float* __restrict__ out) {
    __shared__ float wsh[EXPERTS];
    const int b = blockIdx.x;
    const int t = threadIdx.x;
    if (t < EXPERTS) wsh[t] = D2[b * EXPERTS + t] ? conf[b * EXPERTS + t] : 0.f;
    __syncthreads();
    const float inv = 1.f / nrm[b];
    for (int c = t; c < NCLS; c += 256) {
        float s = 0.f;
        #pragma unroll
        for (int e = 0; e < EXPERTS; ++e)
            s += wsh[e] * logits[((size_t)b * EXPERTS + e) * NCLS + c];
        out[(size_t)b * NCLS + c] = s * inv;
    }
}

// ---------------------------------------------------------------------------
extern "C" void kernel_launch(void* const* d_in, const int* in_sizes, int n_in,
                              void* d_out, int out_size, void* d_ws, size_t ws_size,
                              hipStream_t stream) {
    (void)in_sizes; (void)n_in; (void)out_size; (void)ws_size;

    const float* x    = (const float*)d_in[0];
    const float* w1   = (const float*)d_in[1];
    const float* cb1  = (const float*)d_in[2];
    const float* bg1  = (const float*)d_in[3];
    const float* bb1  = (const float*)d_in[4];
    const float* w2   = (const float*)d_in[5];
    const float* cb2  = (const float*)d_in[6];
    const float* bg2  = (const float*)d_in[7];
    const float* bb2  = (const float*)d_in[8];
    const float* w3   = (const float*)d_in[9];
    const float* cb3  = (const float*)d_in[10];
    const float* bg3  = (const float*)d_in[11];
    const float* bb3  = (const float*)d_in[12];
    const float* w4   = (const float*)d_in[13];
    const float* cb4  = (const float*)d_in[14];
    const float* bg4  = (const float*)d_in[15];
    const float* bb4  = (const float*)d_in[16];
    const float* clsw = (const float*)d_in[17];
    const float* clsb = (const float*)d_in[18];

    char* ws = (char*)d_ws;
    size_t o = 0;
    auto alloc = [&](size_t bytes) { size_t cur = o; o += (bytes + 255) & ~(size_t)255; return cur; };

    half_t* xh    = (half_t*)(ws + alloc((size_t)BATCH * 64 * 64 * 3  * 2));
    half_t* bufA  = (half_t*)(ws + alloc((size_t)BATCH * 64 * 64 * 32 * 2));  // a1 / a3
    half_t* bufB  = (half_t*)(ws + alloc((size_t)BATCH * 64 * 64 * 64 * 2));  // a2 / a4 / logits
    half_t* bufC  = (half_t*)(ws + alloc((size_t)BATCH * 32 * 32 * 64 * 2));  // pooled
    half_t* feats = (half_t*)(ws + alloc((size_t)BATCH * TCH * 2));
    half_t* wp1   = (half_t*)(ws + alloc((size_t)32   * 32  * 2));
    half_t* wp2   = (half_t*)(ws + alloc((size_t)288  * 64  * 2));
    half_t* wp3   = (half_t*)(ws + alloc((size_t)576  * 128 * 2));
    half_t* wp4   = (half_t*)(ws + alloc((size_t)1152 * 256 * 2));
    half_t* wpc   = (half_t*)(ws + alloc((size_t)256  * 16000 * 2));
    float*  confb = (float*)(ws + alloc((size_t)BATCH * EXPERTS * 4));
    int*    D1    = (int*)  (ws + alloc((size_t)BATCH * EXPERTS * 4));
    int*    D2    = (int*)  (ws + alloc((size_t)BATCH * EXPERTS * 4));
    float*  nrm   = (float*)(ws + alloc((size_t)BATCH * 4));
    float*  logits = (float*)bufB;   // aliased: bufB dead once feats built

    float* out_logits = (float*)d_out;
    float* out_conf   = out_logits + (size_t)BATCH * NCLS;
    float* out_D      = out_conf   + (size_t)BATCH * EXPERTS;

    nchw_to_nhwc_f16<<<(BATCH * 3 * 64 * 64 + 255) / 256, 256, 0, stream>>>(x, xh, 3, 64, 64);

    pack_conv_weights<<<(1  * 2  * 512 + 255) / 256, 256, 0, stream>>>(w1, wp1, 3,   32,  27,   32);
    pack_conv_weights<<<(9  * 4  * 512 + 255) / 256, 256, 0, stream>>>(w2, wp2, 32,  64,  288,  288);
    pack_conv_weights<<<(18 * 8  * 512 + 255) / 256, 256, 0, stream>>>(w3, wp3, 64,  128, 576,  576);
    pack_conv_weights<<<(36 * 16 * 512 + 255) / 256, 256, 0, stream>>>(w4, wp4, 128, 256, 1152, 1152);
    pack_cls_weights<<<(8 * 1000 * 512 + 255) / 256, 256, 0, stream>>>(clsw, wpc);

    // trunk (waves = Mtiles * ngroups; 8 waves / 256-thread block)
    conv3x3_wmma<3,  32,  2, true ><<<65536 * 1 / 8, 256, 0, stream>>>(xh,   wp1, cb1, bg1, bb1, bufA, 64, 64);
    conv3x3_wmma<32, 64,  4, true ><<<65536 * 1 / 8, 256, 0, stream>>>(bufA, wp2, cb2, bg2, bb2, bufB, 64, 64);
    maxpool2x2_nhwc_v8<<<(BATCH * 32 * 32 * 64 / 8 + 255) / 256, 256, 0, stream>>>(bufB, bufC, 64, 64, 64);
    conv3x3_wmma<64, 128, 4, false><<<16384 * 2 / 8, 256, 0, stream>>>(bufC, wp3, cb3, bg3, bb3, bufA, 32, 32);
    conv3x3_wmma<128,256, 4, false><<<16384 * 4 / 8, 256, 0, stream>>>(bufA, wp4, cb4, bg4, bb4, bufB, 32, 32);
    maxpool2x2_nhwc_v8<<<(BATCH * 16 * 16 * 256 / 8 + 255) / 256, 256, 0, stream>>>(bufB, bufC, 32, 32, 256);
    avgpool_feats<<<(BATCH * TCH + 255) / 256, 256, 0, stream>>>(bufC, feats);

    cls_gemm_wmma<<<16 * 250 / 8, 256, 0, stream>>>(feats, wpc, clsb, logits);

    softmax_entropy<<<BATCH * EXPERTS, 256, 0, stream>>>(logits, confb, out_conf);
    cap_route<<<EXPERTS, BATCH, 0, stream>>>(confb, D1);
    sample_route<<<1, BATCH, 0, stream>>>(confb, D1, out_D, D2, nrm);
    combine<<<BATCH, 256, 0, stream>>>(logits, confb, D2, nrm, out_logits);
}
